// EMTransformerDecoder_7533372637374
// MI455X (gfx1250) — matrix operand order repaired and checked by reading, not compiled, not run on hardware
//
#include <hip/hip_runtime.h>
#include <hip/hip_bf16.h>
#include <math.h>

// ---------------------------------------------------------------------------
// Problem constants (from reference)
// ---------------------------------------------------------------------------
constexpr int kB   = 4;
constexpr int kH   = 128;
constexpr int kW   = 128;
constexpr int kD   = 512;
constexpr int kNH  = 8;
constexpr int kHD  = 64;     // kD / kNH
constexpr int kDFF = 2048;
constexpr int kL   = 6;
constexpr int kP   = 5;
constexpr int kPP  = 25;     // P*P
constexpr int kNQ  = 1024;
constexpr int kN   = 4096;   // B * NQ
constexpr int kCH  = 512;    // cross-attention query chunk

typedef float  v8f   __attribute__((ext_vector_type(8)));
typedef __bf16 v16bf __attribute__((ext_vector_type(16)));

__device__ __forceinline__ unsigned short f2bf(float f) {
    union { float f; unsigned u; } v; v.f = f;
    unsigned u = v.u;
    unsigned r = (u + 0x7FFFu + ((u >> 16) & 1u)) >> 16;   // RNE
    return (unsigned short)r;
}

// pack two f32 into one dword of two bf16 (round-half-up) with one v_perm_b32
__device__ __forceinline__ unsigned pk2(float lo, float hi) {
    unsigned ulo = __float_as_uint(lo) + 0x8000u;
    unsigned uhi = __float_as_uint(hi) + 0x8000u;
    // result bytes: b0=ulo.b2 b1=ulo.b3 b2=uhi.b2 b3=uhi.b3
    return __builtin_amdgcn_perm(uhi, ulo, 0x07060302u);
}

// ---------------------------------------------------------------------------
// Weight pre-swizzle: f32 [K,Nt] (row-major) -> bf16 fragment order
// dst[(((ktile*ntiles + ntile)*32 + lane)*16 + i)] in this code's contiguous k order (same as the A loads):
//   B fragment: lanes 0-15 -> N=lane,     k = ktile*32 + i      (i=0..15)
//               lanes 16-31 -> N=lane-16, k = ktile*32 + 16 + i
// Each B fragment becomes 32 contiguous bytes -> 2x global_load_b128 in GEMM.
// ---------------------------------------------------------------------------
__global__ void convert_w_kernel(const float* __restrict__ src,
                                 unsigned short* __restrict__ dst,
                                 int K, int Nt, int layers)
{
    size_t tid = (size_t)blockIdx.x * blockDim.x + threadIdx.x;
    size_t per = (size_t)K * Nt;
    if (tid >= per * (size_t)layers) return;
    int    l = (int)(tid / per);
    size_t r = tid % per;
    int i      = (int)(r & 15);
    int lane   = (int)((r >> 4) & 31);
    int ntiles = Nt >> 4;
    int ntile  = (int)((r >> 9) % (size_t)ntiles);
    int ktile  = (int)((r >> 9) / (size_t)ntiles);
    int k   = ktile * 32 + ((lane < 16) ? 0 : 16) + i;
    int col = ntile * 16 + (lane & 15);
    dst[tid] = f2bf(src[(size_t)l * per + (size_t)k * Nt + col]);
}

// ---------------------------------------------------------------------------
// LDS-free WMMA GEMM: C[M,Nt] = act(A[M,K] @ W + bias) (+ residual)
// Block = 256 threads = 8 waves; wave w owns M-tile (blockIdx.x*8+w),
// computing a 16(M) x 64(N) strip with 4 f32 accumulators.
// A fragments: loaded straight from row-major f32 global (two contiguous
// 8-float runs per lane), converted to bf16 in registers via v_perm_b32.
// B fragments: 32B contiguous loads from pre-swizzled bf16 weights.
// No LDS, no barriers; 4 v_wmma_f32_16x16x32_bf16 per K-step.
// act: 0=none, 1=GELU(exact), 2=ReLU.
// ---------------------------------------------------------------------------
__global__ void __launch_bounds__(256)
gemm_wmma(const float* __restrict__ A, const unsigned short* __restrict__ Wbf,
          const float* __restrict__ bias, const float* __restrict__ residual,
          float* __restrict__ C, int M, int Nt, int K, int act)
{
    const int lane  = threadIdx.x & 31;
    const int wave  = threadIdx.x >> 5;
    const int mtile = blockIdx.x * 8 + wave;
    const int mrow  = mtile * 16 + (lane & 15);   // A row this lane streams
    const int n0    = blockIdx.y * 64;
    const int akb   = (lane < 16) ? 0 : 16;       // A k sub-offset per layout
    const int ntiles = Nt >> 4;

    union Acc { v8f v; float e[8]; } acc[4];
    #pragma unroll
    for (int j = 0; j < 4; ++j)
        #pragma unroll
        for (int i = 0; i < 8; ++i) acc[j].e[i] = 0.0f;

    const float* arow = A + (size_t)mrow * K + akb;
    const unsigned short* wbase =
        Wbf + ((size_t)(n0 >> 4) * 32 + lane) * 16;   // ktile 0, first ntile
    const size_t wkstride = (size_t)ntiles * 512;     // ushorts per ktile

    const int kts = K >> 5;
    for (int kt = 0; kt < kts; ++kt) {
        // ---- A fragment: rows contiguous in k; 4x b128 loads ----
        const float4* ap  = (const float4*)(arow + kt * 32);
        const float4* ap2 = (const float4*)(arow + kt * 32 + 8);
        float4 a0 = ap[0],  a1 = ap[1];
        float4 a2 = ap2[0], a3 = ap2[1];
        union { v16bf v; unsigned u[8]; } af;
        af.u[0] = pk2(a0.x, a0.y); af.u[1] = pk2(a0.z, a0.w);
        af.u[2] = pk2(a1.x, a1.y); af.u[3] = pk2(a1.z, a1.w);
        af.u[4] = pk2(a2.x, a2.y); af.u[5] = pk2(a2.z, a2.w);
        af.u[6] = pk2(a3.x, a3.y); af.u[7] = pk2(a3.z, a3.w);

        const unsigned short* wk = wbase + (size_t)kt * wkstride;
        // prefetch next panels (global_prefetch_b8; speculative past-end OK)
        __builtin_prefetch(wk + wkstride, 0, 0);
        __builtin_prefetch(arow + (kt + 1) * 32, 0, 0);

        #pragma unroll
        for (int j = 0; j < 4; ++j) {
            union { v16bf v; uint4 q[2]; } bg;
            const uint4* bp = (const uint4*)(wk + j * 512);
            bg.q[0] = bp[0];
            bg.q[1] = bp[1];
            acc[j].v = __builtin_amdgcn_wmma_f32_16x16x32_bf16(
                false, af.v, false, bg.v, (short)0, acc[j].v, false, false);
        }
    }

    // ---- epilogue: C layout VGPR r -> M=r (lanes 0-15) / M=r+8 (16-31) ----
    const int mb = (lane < 16) ? 0 : 8;
    #pragma unroll
    for (int j = 0; j < 4; ++j) {
        const int ncol = n0 + j * 16 + (lane & 15);
        const float bv = bias[ncol];
        #pragma unroll
        for (int r = 0; r < 8; ++r) {
            int mm = mtile * 16 + r + mb;
            float v = acc[j].e[r] + bv;
            if (act == 1)      v = 0.5f * v * (1.0f + erff(v * 0.70710678118654752f));
            else if (act == 2) v = fmaxf(v, 0.0f);
            if (residual)      v += residual[(size_t)mm * Nt + ncol];
            C[(size_t)mm * Nt + ncol] = v;
        }
    }
}

// ---------------------------------------------------------------------------
// State init: x = queries, pos/sub/idx working copies (inputs stay untouched)
// ---------------------------------------------------------------------------
__global__ void init_state_kernel(const float* __restrict__ q,
                                  const float* __restrict__ pos,
                                  const float* __restrict__ sub,
                                  const int*   __restrict__ idx,
                                  float* __restrict__ x, float* __restrict__ posb,
                                  float* __restrict__ subb, int* __restrict__ idxb)
{
    int tid = blockIdx.x * blockDim.x + threadIdx.x;
    if (tid < kN * kD) x[tid] = q[tid];
    if (tid < kN * 2) { posb[tid] = pos[tid]; subb[tid] = sub[tid]; }
    if (tid < kN * 3) idxb[tid] = idx[tid];
}

// ---------------------------------------------------------------------------
// x += sinusoid(pixel indices) + sinusoid(subpixel)
// ---------------------------------------------------------------------------
__global__ void add_pe_kernel(float* __restrict__ x, const int* __restrict__ idxb,
                              const float* __restrict__ subb)
{
    int tid = blockIdx.x * blockDim.x + threadIdx.x;
    if (tid >= kN * kD) return;
    int n = tid >> 9;
    int d = tid & 511;
    float p1 = (float)idxb[n * 3 + 1];
    float p2 = (float)idxb[n * 3 + 2];
    float s1 = subb[n * 2 + 0];
    float s2 = subb[n * 2 + 1];
    int quad = d >> 7;
    int j    = d & 127;
    float inv = __expf(-(float)j * (9.210340371976184f / 128.0f)); // 10000^-j/128
    float pp = (quad < 2) ? p1 : p2;
    float ss = (quad < 2) ? s1 : s2;
    float pe, se;
    if ((quad & 1) == 0) { pe = sinf(pp * inv); se = sinf(ss * inv); }
    else                 { pe = cosf(pp * inv); se = cosf(ss * inv); }
    x[tid] += pe + se;
}

// ---------------------------------------------------------------------------
// LayerNorm over D=512, one row per block
// ---------------------------------------------------------------------------
__global__ void __launch_bounds__(256)
ln_kernel(const float* __restrict__ x, float* __restrict__ out,
          const float* __restrict__ w, const float* __restrict__ b)
{
    __shared__ float red[256];
    int n = blockIdx.x, t = threadIdx.x;
    const float* row = x + (size_t)n * kD;
    float s = 0.f;
    for (int i = t; i < kD; i += 256) s += row[i];
    red[t] = s; __syncthreads();
    for (int st = 128; st > 0; st >>= 1) {
        if (t < st) red[t] += red[t + st];
        __syncthreads();
    }
    float mean = red[0] / (float)kD;
    __syncthreads();
    float v = 0.f;
    for (int i = t; i < kD; i += 256) { float q = row[i] - mean; v += q * q; }
    red[t] = v; __syncthreads();
    for (int st = 128; st > 0; st >>= 1) {
        if (t < st) red[t] += red[t + st];
        __syncthreads();
    }
    float inv = rsqrtf(red[0] / (float)kD + 1e-5f);
    for (int i = t; i < kD; i += 256)
        out[(size_t)n * kD + i] = (row[i] - mean) * inv * w[i] + b[i];
}

// ---------------------------------------------------------------------------
// Fused self-attention, one (b,h,q) per 128-thread block; 1024 keys in LDS.
// The query head vector is staged into LDS with the CDNA5 async-LDS path
// (global_load_async_to_lds_b128 + s_wait_asynccnt).
// ---------------------------------------------------------------------------
__global__ void __launch_bounds__(128)
self_attn_kernel(const float* __restrict__ qkv, float* __restrict__ o)
{
    __shared__ float qs[kHD];
    __shared__ float sc[kNQ];
    __shared__ float red[128];
    int bid = blockIdx.x;
    int qi = bid & (kNQ - 1);
    int h  = (bid >> 10) & (kNH - 1);
    int b  = bid >> 13;
    int t  = threadIdx.x;
    const size_t stride = 3 * kD;
    const size_t rowq = (size_t)(b * kNQ + qi) * stride + h * kHD;

#if defined(__gfx1250__)
    if (t < kHD / 4) {   // 16 lanes x 16B = 64 floats
        const float* gsrc = qkv + rowq + t * 4;
        unsigned ldsoff = (unsigned)(size_t)&qs[t * 4];  // low 32b = LDS offset
        asm volatile("global_load_async_to_lds_b128 %0, %1, off"
                     :: "v"(ldsoff), "v"(gsrc) : "memory");
    }
    asm volatile("s_wait_asynccnt 0x0" ::: "memory");
#else
    if (t < kHD) qs[t] = qkv[rowq + t];
#endif
    __syncthreads();

    float mx = -3.4e38f;
    for (int k = t; k < kNQ; k += 128) {
        const float* kr = qkv + (size_t)(b * kNQ + k) * stride + kD + h * kHD;
        float dp = 0.f;
        #pragma unroll 8
        for (int j = 0; j < kHD; ++j) dp += qs[j] * kr[j];
        dp *= 0.125f;                       // 1/sqrt(hd)
        sc[k] = dp;
        mx = fmaxf(mx, dp);
    }
    red[t] = mx; __syncthreads();
    for (int st = 64; st > 0; st >>= 1) {
        if (t < st) red[t] = fmaxf(red[t], red[t + st]);
        __syncthreads();
    }
    mx = red[0]; __syncthreads();
    float sum = 0.f;
    for (int k = t; k < kNQ; k += 128) {
        float e = __expf(sc[k] - mx);
        sc[k] = e; sum += e;
    }
    red[t] = sum; __syncthreads();
    for (int st = 64; st > 0; st >>= 1) {
        if (t < st) red[t] += red[t + st];
        __syncthreads();
    }
    float inv = 1.f / red[0];
    __syncthreads();

    int dd = t & 63, half = t >> 6;
    float acc = 0.f;
    for (int k = half * 512; k < (half + 1) * 512; ++k) {
        const float* vr = qkv + (size_t)(b * kNQ + k) * stride + 2 * kD + h * kHD;
        acc += sc[k] * vr[dd];
    }
    red[t] = acc; __syncthreads();
    if (t < kHD)
        o[(size_t)(b * kNQ + qi) * kD + h * kHD + t] = (red[t] + red[t + 64]) * inv;
}

// ---------------------------------------------------------------------------
// Cross-attn key gather + bilinear table encoding, one element per thread
// ---------------------------------------------------------------------------
__global__ void gather_kernel(const float* __restrict__ img, const float* __restrict__ tab,
                              const int* __restrict__ idxb, const float* __restrict__ subb,
                              int n0, float* __restrict__ keys, float* __restrict__ kin)
{
    int tid = blockIdx.x * blockDim.x + threadIdx.x;
    if (tid >= kCH * kPP * kD) return;
    int d  = tid & 511;
    int p  = (tid >> 9) % kPP;
    int nl = tid / (kPP * 512);
    int n  = n0 + nl;
    int oi = p / kP - kP / 2;
    int oj = p % kP - kP / 2;
    int bq = idxb[n * 3 + 0];
    int i1 = idxb[n * 3 + 1];
    int i2 = idxb[n * 3 + 2];
    int ri = min(max(i1 + oi, 0), kH - 1);
    int cj = min(max(i2 + oj, 0), kW - 1);
    float kv = img[(((size_t)bq * kH + ri) * kW + cj) * kD + d];

    float s0 = subb[n * 2 + 0], s1 = subb[n * 2 + 1];
    float ti = (float)oi + 0.5f - s0 + (float)kP * 0.5f;
    float tj = (float)oj + 0.5f - s1 + (float)kP * 0.5f;
    float fi0 = fminf(fmaxf(floorf(ti), 0.f), (float)(kP - 1));
    float fj0 = fminf(fmaxf(floorf(tj), 0.f), (float)(kP - 1));
    int   i0  = (int)fi0;  float fi = ti - fi0;
    int   j0  = (int)fj0;  float fj = tj - fj0;
    const int TS = kP + 1;
    const float* t00 = tab + (size_t)(i0 * TS + j0) * kD;
    const float* t10 = tab + (size_t)((i0 + 1) * TS + j0) * kD;
    const float* t01 = tab + (size_t)(i0 * TS + j0 + 1) * kD;
    const float* t11 = tab + (size_t)((i0 + 1) * TS + j0 + 1) * kD;
    float enc = (1.f - fi) * (1.f - fj) * t00[d] + fi * (1.f - fj) * t10[d]
              + (1.f - fi) * fj * t01[d] + fi * fj * t11[d];

    size_t out = ((size_t)nl * kPP + p) * kD + d;
    keys[out] = kv;
    kin[out]  = kv + enc;
}

// ---------------------------------------------------------------------------
// Fused 25-key cross-attention, one query per 256-thread block
// ---------------------------------------------------------------------------
__global__ void __launch_bounds__(256)
cross_attn_kernel(const float* __restrict__ qp, const float* __restrict__ kp,
                  const float* __restrict__ vp, const int* __restrict__ idxb,
                  int n0, float* __restrict__ o)
{
    __shared__ float qs[kD];
    __shared__ float sc[kNH][kPP];
    int nl = blockIdx.x;
    int n  = n0 + nl;
    int t  = threadIdx.x;

    for (int i = t; i < kD; i += 256) qs[i] = qp[(size_t)n * kD + i] * 0.125f;
    __syncthreads();

    if (t < kNH * kPP) {
        int h = t / kPP, kk = t % kPP;
        int oi = kk / kP - kP / 2, oj = kk % kP - kP / 2;
        int i1 = idxb[n * 3 + 1], i2 = idxb[n * 3 + 2];
        bool valid = (i1 + oi >= 0) && (i1 + oi < kH) && (i2 + oj >= 0) && (i2 + oj < kW);
        const float* kr = kp + ((size_t)nl * kPP + kk) * kD + h * kHD;
        float dp = 0.f;
        #pragma unroll 8
        for (int j = 0; j < kHD; ++j) dp += qs[h * kHD + j] * kr[j];
        sc[h][kk] = valid ? dp : -3.40282347e38f;
    }
    __syncthreads();
    if (t < kNH) {
        float mx = -3.4e38f;
        for (int k = 0; k < kPP; ++k) mx = fmaxf(mx, sc[t][k]);
        float sum = 0.f;
        for (int k = 0; k < kPP; ++k) { float e = __expf(sc[t][k] - mx); sc[t][k] = e; sum += e; }
        float inv = 1.f / sum;
        for (int k = 0; k < kPP; ++k) sc[t][k] *= inv;
    }
    __syncthreads();
    for (int dd = t; dd < kD; dd += 256) {
        int h = dd >> 6;
        float acc = 0.f;
        #pragma unroll
        for (int k = 0; k < kPP; ++k)
            acc += sc[h][k] * vp[((size_t)nl * kPP + k) * kD + dd];
        o[(size_t)n * kD + dd] = acc;
    }
}

// ---------------------------------------------------------------------------
// delta = tanh(relu_feats @ d_w2 + d_b2); update pos/sub/idx working state
// ---------------------------------------------------------------------------
__global__ void __launch_bounds__(128)
delta_kernel(const float* __restrict__ r, const float* __restrict__ w2,
             const float* __restrict__ b2, float* __restrict__ posb,
             float* __restrict__ subb, int* __restrict__ idxb)
{
    __shared__ float red0[128], red1[128];
    int n = blockIdx.x, t = threadIdx.x;
    float a0 = 0.f, a1 = 0.f;
    for (int j = t; j < kD; j += 128) {
        float rv = r[(size_t)n * kD + j];
        a0 += rv * w2[j * 2 + 0];
        a1 += rv * w2[j * 2 + 1];
    }
    red0[t] = a0; red1[t] = a1; __syncthreads();
    for (int st = 64; st > 0; st >>= 1) {
        if (t < st) { red0[t] += red0[t + st]; red1[t] += red1[t + st]; }
        __syncthreads();
    }
    if (t == 0) {
        float d0 = tanhf(red0[0] + b2[0]);
        float d1 = tanhf(red1[0] + b2[1]);
        float p0 = fminf(fmaxf(posb[n * 2 + 0] + d0, 0.f), (float)kH);
        float p1 = fminf(fmaxf(posb[n * 2 + 1] + d1, 0.f), (float)kW);
        posb[n * 2 + 0] = p0; posb[n * 2 + 1] = p1;
        float x0 = floorf(p0), x1 = floorf(p1);
        subb[n * 2 + 0] = p0 - x0; subb[n * 2 + 1] = p1 - x1;
        idxb[n * 3 + 1] = (int)x0; idxb[n * 3 + 2] = (int)x1;
    }
}

__global__ void copy_kernel(const float* __restrict__ src, float* __restrict__ dst, int cnt)
{
    int tid = blockIdx.x * blockDim.x + threadIdx.x;
    if (tid < cnt) dst[tid] = src[tid];
}

// ---------------------------------------------------------------------------
// Host orchestration (all launches on `stream`; graph-capture-safe)
// ---------------------------------------------------------------------------
extern "C" void kernel_launch(void* const* d_in, const int* in_sizes, int n_in,
                              void* d_out, int out_size, void* d_ws, size_t ws_size,
                              hipStream_t stream)
{
    (void)in_sizes; (void)n_in; (void)out_size; (void)ws_size;

    const float* queries = (const float*)d_in[0];
    const float* pos_in  = (const float*)d_in[1];
    const float* sub_in  = (const float*)d_in[2];
    const int*   idx_in  = (const int*)  d_in[3];
    /* d_in[4] = batch_offsets (unused by reference) */
    const float* img     = (const float*)d_in[5];
    const float* sa_in_w  = (const float*)d_in[6];
    const float* sa_in_b  = (const float*)d_in[7];
    const float* sa_out_w = (const float*)d_in[8];
    const float* sa_out_b = (const float*)d_in[9];
    const float* sa_ln_w  = (const float*)d_in[10];
    const float* sa_ln_b  = (const float*)d_in[11];
    const float* ca_q_w   = (const float*)d_in[12];
    const float* ca_q_b   = (const float*)d_in[13];
    const float* ca_k_w   = (const float*)d_in[14];
    const float* ca_k_b   = (const float*)d_in[15];
    const float* ca_v_w   = (const float*)d_in[16];
    const float* ca_v_b   = (const float*)d_in[17];
    const float* ca_o_w   = (const float*)d_in[18];
    const float* ca_o_b   = (const float*)d_in[19];
    const float* ca_tab   = (const float*)d_in[20];
    const float* f_w1     = (const float*)d_in[21];
    const float* f_b1     = (const float*)d_in[22];
    const float* f_w2     = (const float*)d_in[23];
    const float* f_b2     = (const float*)d_in[24];
    const float* f_ln_w   = (const float*)d_in[25];
    const float* f_ln_b   = (const float*)d_in[26];
    const float* d_w1     = (const float*)d_in[27];
    const float* d_b1     = (const float*)d_in[28];
    const float* d_w2     = (const float*)d_in[29];
    const float* d_b2     = (const float*)d_in[30];

    // ---- workspace carve-up ----
    float* ws = (float*)d_ws;
    size_t off = 0;
    auto alloc   = [&](size_t cnt) { float* p = ws + off; off += cnt; return p; };
    auto allocUS = [&](size_t cnt) {
        unsigned short* p = (unsigned short*)(ws + off);
        off += (cnt + 1) / 2;
        return p;
    };
    float* xbuf   = alloc((size_t)kN * kD);
    float* hbuf   = alloc((size_t)kN * kD);
    float* qkvbuf = alloc((size_t)kN * 3 * kD);
    float* obuf   = alloc((size_t)kN * kD);
    float* qproj  = alloc((size_t)kN * kD);
    float* keysb  = alloc((size_t)kCH * kPP * kD);
    float* kinb   = alloc((size_t)kCH * kPP * kD);
    float* kprojb = alloc((size_t)kCH * kPP * kD);
    float* vprojb = alloc((size_t)kCH * kPP * kD);
    float* ffnb   = alloc((size_t)kN * kDFF);
    float* rbuf   = alloc((size_t)kN * kD);
    float* posb   = alloc((size_t)kN * 2);
    float* subb   = alloc((size_t)kN * 2);
    int*   idxb   = (int*)alloc((size_t)kN * 3 + 1);
    // pre-swizzled bf16 weights (fragment order)
    unsigned short* w_sa_in  = allocUS((size_t)kL * kD * 3 * kD);
    unsigned short* w_sa_out = allocUS((size_t)kL * kD * kD);
    unsigned short* w_ca_q   = allocUS((size_t)kL * kD * kD);
    unsigned short* w_ca_k   = allocUS((size_t)kL * kD * kD);
    unsigned short* w_ca_v   = allocUS((size_t)kL * kD * kD);
    unsigned short* w_ca_o   = allocUS((size_t)kL * kD * kD);
    unsigned short* w_f1     = allocUS((size_t)kL * kD * kDFF);
    unsigned short* w_f2     = allocUS((size_t)kL * kDFF * kD);
    unsigned short* w_d1     = allocUS((size_t)kD * kD);

    const int TPB = 256;
    auto convw = [&](const float* src, unsigned short* dst, int K_, int Nt_, int layers_) {
        size_t tot = (size_t)K_ * Nt_ * layers_;
        convert_w_kernel<<<(unsigned)((tot + TPB - 1) / TPB), TPB, 0, stream>>>(
            src, dst, K_, Nt_, layers_);
    };
    convw(sa_in_w,  w_sa_in,  kD, 3 * kD, kL);
    convw(sa_out_w, w_sa_out, kD, kD, kL);
    convw(ca_q_w,   w_ca_q,   kD, kD, kL);
    convw(ca_k_w,   w_ca_k,   kD, kD, kL);
    convw(ca_v_w,   w_ca_v,   kD, kD, kL);
    convw(ca_o_w,   w_ca_o,   kD, kD, kL);
    convw(f_w1,     w_f1,     kD, kDFF, kL);
    convw(f_w2,     w_f2,     kDFF, kD, kL);
    convw(d_w1,     w_d1,     kD, kD, 1);

    init_state_kernel<<<(kN * kD + TPB - 1) / TPB, TPB, 0, stream>>>(
        queries, pos_in, sub_in, idx_in, xbuf, posb, subb, idxb);

    for (int l = 0; l < kL; ++l) {
        const unsigned short* siw = w_sa_in  + (size_t)l * kD * 3 * kD;
        const unsigned short* sow = w_sa_out + (size_t)l * kD * kD;
        const unsigned short* cqw = w_ca_q   + (size_t)l * kD * kD;
        const unsigned short* ckw = w_ca_k   + (size_t)l * kD * kD;
        const unsigned short* cvw = w_ca_v   + (size_t)l * kD * kD;
        const unsigned short* cow = w_ca_o   + (size_t)l * kD * kD;
        const unsigned short* fw1 = w_f1     + (size_t)l * kD * kDFF;
        const unsigned short* fw2 = w_f2     + (size_t)l * kDFF * kD;
        const float* sib = sa_in_b  + (size_t)l * 3 * kD;
        const float* sob = sa_out_b + (size_t)l * kD;
        const float* slw = sa_ln_w  + (size_t)l * kD;
        const float* slb = sa_ln_b  + (size_t)l * kD;
        const float* cqb = ca_q_b   + (size_t)l * kD;
        const float* ckb = ca_k_b   + (size_t)l * kD;
        const float* cvb = ca_v_b   + (size_t)l * kD;
        const float* cob = ca_o_b   + (size_t)l * kD;
        const float* tab = ca_tab   + (size_t)l * (kP + 1) * (kP + 1) * kD;
        const float* fb1 = f_b1     + (size_t)l * kDFF;
        const float* fb2 = f_b2     + (size_t)l * kD;
        const float* flw = f_ln_w   + (size_t)l * kD;
        const float* flb = f_ln_b   + (size_t)l * kD;

        add_pe_kernel<<<(kN * kD + TPB - 1) / TPB, TPB, 0, stream>>>(xbuf, idxb, subb);

        // ---- self-attention ----
        ln_kernel<<<kN, 256, 0, stream>>>(xbuf, hbuf, slw, slb);
        gemm_wmma<<<dim3(kN / 128, (3 * kD) / 64), 256, 0, stream>>>(
            hbuf, siw, sib, nullptr, qkvbuf, kN, 3 * kD, kD, 0);
        self_attn_kernel<<<kB * kNH * kNQ, 128, 0, stream>>>(qkvbuf, obuf);
        gemm_wmma<<<dim3(kN / 128, kD / 64), 256, 0, stream>>>(
            obuf, sow, sob, xbuf, xbuf, kN, kD, kD, 0);

        // ---- cross-attention ----
        gemm_wmma<<<dim3(kN / 128, kD / 64), 256, 0, stream>>>(
            xbuf, cqw, cqb, nullptr, qproj, kN, kD, kD, 0);
        for (int c = 0; c < kN / kCH; ++c) {
            int n0 = c * kCH;
            gather_kernel<<<(kCH * kPP * kD + TPB - 1) / TPB, TPB, 0, stream>>>(
                img, tab, idxb, subb, n0, keysb, kinb);
            gemm_wmma<<<dim3((kCH * kPP) / 128, kD / 64), 256, 0, stream>>>(
                kinb, ckw, ckb, nullptr, kprojb, kCH * kPP, kD, kD, 0);
            gemm_wmma<<<dim3((kCH * kPP) / 128, kD / 64), 256, 0, stream>>>(
                keysb, cvw, cvb, nullptr, vprojb, kCH * kPP, kD, kD, 0);
            cross_attn_kernel<<<kCH, 256, 0, stream>>>(
                qproj, kprojb, vprojb, idxb, n0, obuf);
        }
        gemm_wmma<<<dim3(kN / 128, kD / 64), 256, 0, stream>>>(
            obuf, cow, cob, xbuf, xbuf, kN, kD, kD, 0);

        // ---- FFN ----
        ln_kernel<<<kN, 256, 0, stream>>>(xbuf, hbuf, flw, flb);
        gemm_wmma<<<dim3(kN / 128, kDFF / 64), 256, 0, stream>>>(
            hbuf, fw1, fb1, nullptr, ffnb, kN, kDFF, kD, 1 /*GELU*/);
        gemm_wmma<<<dim3(kN / 128, kD / 64), 256, 0, stream>>>(
            ffnb, fw2, fb2, xbuf, xbuf, kN, kD, kDFF, 0);

        // ---- position update head ----
        gemm_wmma<<<dim3(kN / 128, kD / 64), 256, 0, stream>>>(
            xbuf, w_d1, d_b1, nullptr, rbuf, kN, kD, kD, 2 /*ReLU*/);
        delta_kernel<<<kN, 128, 0, stream>>>(rbuf, d_w2, d_b2, posb, subb, idxb);
    }

    copy_kernel<<<(kN * kD + TPB - 1) / TPB, TPB, 0, stream>>>(xbuf, (float*)d_out, kN * kD);
}
